// LSTMNetwork_49959059587099
// MI455X (gfx1250) — compile-verified, matrix operand order checked by reading, not run on hardware
//
#include <hip/hip_runtime.h>
#include <hip/hip_bf16.h>

typedef __attribute__((ext_vector_type(16))) _Float16 v16h;
typedef __attribute__((ext_vector_type(8)))  _Float16 v8h;
typedef __attribute__((ext_vector_type(8)))  float    v8f;
typedef __attribute__((ext_vector_type(4)))  int      v4i;

#define AS1 __attribute__((address_space(1)))
#define AS3 __attribute__((address_space(3)))

#if defined(__has_builtin)
#if __has_builtin(__builtin_amdgcn_global_load_async_to_lds_b128) && \
    __has_builtin(__builtin_amdgcn_s_wait_asynccnt)
#define USE_ASYNC_LDS 1
#endif
#endif

#define B_  64
#define T_  2048
#define D_  128
#define H_  128
#define G_  512     // 4*H
#define O_  32
#define SP  136     // LDS row stride in halfs (272B; 16B-aligned rows)
#define XQH (16 * SP)           // halfs per x/h tile buffer (2176)
#define WHH_HALFS (G_ * H_)     // 65536 halfs = 128 KB packed w_hh
#define XQ0 (WHH_HALFS)             // offset of xq double buffer (halfs)
#define HQ0 (WHH_HALFS + 2 * XQH)   // offset of hq double buffer (halfs)
#define SMEM_BYTES ((WHH_HALFS + 4 * XQH) * 2)   // 148480 B dynamic LDS

__device__ __forceinline__ float quantf(float x) {
  // round-to-nearest-even at 2^-8, clip to int16 range, back to float
  float q = rintf(x * 256.0f);
  q = fminf(fmaxf(q, -32768.0f), 32767.0f);
  return q * 0.00390625f;
}
__device__ __forceinline__ float hsig(float x) {
  return fminf(fmaxf(x * 0.16666667f + 0.5f, 0.0f), 1.0f);
}
__device__ __forceinline__ float htanh(float x) {
  return fminf(fmaxf(x, -1.0f), 1.0f);
}

// ---------------------------------------------------------------------------
// Phase 0: quantize w_ih / w_hh to f16 and pack into WMMA B-fragment layout:
//   pack[which][n_tile(32)][k_tile(4)][lane(32)][16 halfs]
// Lane L holds column n = L%16; halfs 0..7 cover K = (L/16)*8 + 0..7,
// halfs 8..15 cover K = 16 + (L/16)*8 + 0..7 (16-bit B layout, K=32 tile).
// ---------------------------------------------------------------------------
__global__ void pack_weights_kernel(const float* __restrict__ w_ih,
                                    const float* __restrict__ w_hh,
                                    _Float16* __restrict__ pack) {
  int i = blockIdx.x * blockDim.x + threadIdx.x;   // 0 .. 131071
  int which = i >> 16;                             // 0: w_ih, 1: w_hh
  int r  = i & 0xFFFF;
  int h  = r & 15;
  int L  = (r >> 4) & 31;
  int kt = (r >> 9) & 3;
  int nt = r >> 11;                                // 0..31
  int n  = L & 15, khalf = L >> 4;
  int kl = (h < 8) ? (khalf * 8 + h) : (16 + khalf * 8 + (h - 8));
  int kg = kt * 32 + kl;
  int ng = nt * 16 + n;
  const float* W = which ? w_hh : w_ih;
  pack[i] = (_Float16)quantf(W[ng * D_ + kg]);
}

// ---------------------------------------------------------------------------
// Persistent fused LSTM scan. Grid = 4 blocks x 256 threads (8 waves).
// Block bg owns batch rows [bg*16, bg*16+16). Wave w owns H-columns
// [w*16, w*16+16): its 4 gate tiles are i/f/g/o for those columns and its
// c-state tile (16x16 f32) stays in 8 VGPRs for the entire scan.
// w_hh (packed, 128 KB) is staged to LDS once (async-to-LDS when available).
// xq / hq are double-buffered BY INTEGER OFFSET into one __shared__ array so
// address-space inference keeps every access on the ds_ path, and a single
// barrier per timestep suffices.
// ---------------------------------------------------------------------------
__global__ void __launch_bounds__(256)
lstm_scan_kernel(const float* __restrict__ x,
                 const float* __restrict__ b_ih,
                 const float* __restrict__ b_hh,
                 const float* __restrict__ w_out,
                 const float* __restrict__ b_out,
                 const _Float16* __restrict__ pack,
                 float* __restrict__ out) {
  extern __shared__ _Float16 smem[];
  // layout (halfs): [0, WHH_HALFS) packed w_hh | xq0 | xq1 | hq0 | hq1

  const int tid  = threadIdx.x;
  const int lane = tid & 31;
  const int wv   = tid >> 5;        // wave id 0..7
  const int n    = lane & 15;       // lane%16: A-row / C-column index
  const int kh   = lane >> 4;       // lane/16: K-half select
  const int bg   = blockIdx.x;      // batch-row group

  const _Float16* Wih = pack;
  const _Float16* Whh = pack + (G_ * D_);

  // ---- stage packed w_hh into LDS (one time) ----
#if USE_ASYNC_LDS
#pragma unroll
  for (int j = 0; j < WHH_HALFS / (256 * 8); ++j) {       // 32 iters
    int off = (j * 256 + tid) * 8;                        // coalesced 16B/lane
    __builtin_amdgcn_global_load_async_to_lds_b128(
        (AS1 v4i*)(Whh + off), (AS3 v4i*)(smem + off), 0, 0);
  }
  __builtin_amdgcn_s_wait_asynccnt(0);
#else
  for (int j = 0; j < WHH_HALFS / (256 * 8); ++j) {
    int off = (j * 256 + tid) * 8;
    *(v8h*)(smem + off) = *(const v8h*)(Whh + off);
  }
#endif

  // ---- init h buffers to zero (h0 = 0) ----
  for (int i = tid; i < 2 * XQH; i += 256) smem[HQ0 + i] = (_Float16)0.0f;

  // quantized biases for this wave's 4 gate tiles (per C-layout column)
  float bihq[4], bhhq[4];
#pragma unroll
  for (int j = 0; j < 4; ++j) {
    int col = j * H_ + wv * 16 + n;
    bihq[j] = quantf(b_ih[col]);
    bhhq[j] = quantf(b_hh[col]);
  }

  float cst[8];
#pragma unroll
  for (int e = 0; e < 8; ++e) cst[e] = 0.0f;

  // cooperative x loader mapping: thread -> (row, 8 contiguous cols)
  const int xrow = tid >> 4;
  const int xcol = (tid & 15) * 8;
  const float* xbase = x + ((size_t)(bg * 16 + xrow) * T_) * (size_t)D_ + xcol;

  // software-pipelined x: registers hold x_t while computing step t-1
  float4 a0 = *(const float4*)(xbase);
  float4 a1 = *(const float4*)(xbase + 4);

  for (int t = 0; t < T_; ++t) {
    const int cur = t & 1;
    const int xqo = XQ0 + cur * XQH;          // x_t buffer (read this step)
    const int hqo = HQ0 + cur * XQH;          // h_t buffer (read this step)
    const int hno = HQ0 + (cur ^ 1) * XQH;    // h_{t+1} buffer (written)

    // ---- commit quantized x_t to LDS, then start fetching x_{t+1} ----
    v8h xv;
    xv[0] = (_Float16)quantf(a0.x); xv[1] = (_Float16)quantf(a0.y);
    xv[2] = (_Float16)quantf(a0.z); xv[3] = (_Float16)quantf(a0.w);
    xv[4] = (_Float16)quantf(a1.x); xv[5] = (_Float16)quantf(a1.y);
    xv[6] = (_Float16)quantf(a1.z); xv[7] = (_Float16)quantf(a1.w);
    *(v8h*)(&smem[xqo + xrow * SP + xcol]) = xv;
    if (t + 1 < T_) {
      const float* xp = xbase + (size_t)(t + 1) * D_;
      a0 = *(const float4*)(xp);
      a1 = *(const float4*)(xp + 4);
      __builtin_prefetch(xp + D_, 0, 0);   // global_prefetch_b8 for x_{t+2}
    }
    __syncthreads();   // single barrier per step (double-buffered xq/hq)

    // ---- gates = quant(quant(x@Wih+bih) + quant(h@Whh+bhh)) ----
    float gac[4][8];
#pragma unroll
    for (int j = 0; j < 4; ++j) {
      const int nt = j * 8 + wv;    // gate j, this wave's column block
      v8f p  = {0.f,0.f,0.f,0.f,0.f,0.f,0.f,0.f};
      v8f rr = {0.f,0.f,0.f,0.f,0.f,0.f,0.f,0.f};
#pragma unroll
      for (int kt = 0; kt < 4; ++kt) {
        v8h alo = *(const v8h*)(&smem[xqo + n * SP + kt * 32 + kh * 8]);
        v8h ahi = *(const v8h*)(&smem[xqo + n * SP + kt * 32 + 16 + kh * 8]);
        v16h a = __builtin_shufflevector(alo, ahi, 0,1,2,3,4,5,6,7,8,9,10,11,12,13,14,15);
        const _Float16* bp = Wih + (((nt * 4 + kt) * 32 + lane) << 4);
        v8h blo = *(const v8h*)(bp);
        v8h bhi = *(const v8h*)(bp + 8);
        v16h b = __builtin_shufflevector(blo, bhi, 0,1,2,3,4,5,6,7,8,9,10,11,12,13,14,15);
        p = __builtin_amdgcn_wmma_f32_16x16x32_f16(false, a, false, b, (short)0, p, false, false);
      }
#pragma unroll
      for (int kt = 0; kt < 4; ++kt) {
        v8h alo = *(const v8h*)(&smem[hqo + n * SP + kt * 32 + kh * 8]);
        v8h ahi = *(const v8h*)(&smem[hqo + n * SP + kt * 32 + 16 + kh * 8]);
        v16h a = __builtin_shufflevector(alo, ahi, 0,1,2,3,4,5,6,7,8,9,10,11,12,13,14,15);
        const _Float16* bp = &smem[((nt * 4 + kt) * 32 + lane) << 4];  // LDS B-frag
        v8h blo = *(const v8h*)(bp);
        v8h bhi = *(const v8h*)(bp + 8);
        v16h b = __builtin_shufflevector(blo, bhi, 0,1,2,3,4,5,6,7,8,9,10,11,12,13,14,15);
        rr = __builtin_amdgcn_wmma_f32_16x16x32_f16(false, a, false, b, (short)0, rr, false, false);
      }
#pragma unroll
      for (int e = 0; e < 8; ++e) {
        float pre = quantf(p[e]  + bihq[j]);
        float rec = quantf(rr[e] + bhhq[j]);
        gac[j][e] = quantf(pre + rec);
      }
    }

    // ---- elementwise LSTM cell update (C-layout: row m=e+8*kh, col n) ----
#pragma unroll
    for (int e = 0; e < 8; ++e) {
      float iv = quantf(hsig(gac[0][e]));
      float fv = quantf(hsig(gac[1][e]));
      float gv = quantf(htanh(gac[2][e]));
      float ov = quantf(hsig(gac[3][e]));
      float cv = quantf(quantf(fv * cst[e]) + quantf(iv * gv));
      cst[e] = cv;
      float hv = quantf(ov * quantf(htanh(cv)));
      int m = e + 8 * kh;
      smem[hno + m * SP + wv * 16 + n] = (_Float16)hv;   // next step's h
    }
  }
  __syncthreads();   // h_last (in hq buffer T_&1 = 0) visible to all waves

  // ---- output head: out = quant(h_last @ quant(w_out).T + quant(b_out)) ----
  const int hlo = HQ0 + (T_ & 1) * XQH;
  for (int idx = tid; idx < 16 * O_; idx += 256) {
    int m = idx & 15;
    int o = idx >> 4;
    float acc = quantf(b_out[o]);
    const float* wr = w_out + o * H_;
#pragma unroll 8
    for (int k = 0; k < H_; ++k)
      acc += (float)smem[hlo + m * SP + k] * quantf(wr[k]);
    out[(bg * 16 + m) * O_ + o] = quantf(acc);
  }
}

extern "C" void kernel_launch(void* const* d_in, const int* in_sizes, int n_in,
                              void* d_out, int out_size, void* d_ws, size_t ws_size,
                              hipStream_t stream) {
  const float* x     = (const float*)d_in[0];
  const float* w_ih  = (const float*)d_in[1];
  const float* w_hh  = (const float*)d_in[2];
  const float* b_ih  = (const float*)d_in[3];
  const float* b_hh  = (const float*)d_in[4];
  const float* w_out = (const float*)d_in[5];
  const float* b_out = (const float*)d_in[6];
  float* out = (float*)d_out;

  _Float16* pack = (_Float16*)d_ws;   // 2 * 512*128 halfs = 256 KB

  pack_weights_kernel<<<(2 * G_ * D_) / 256, 256, 0, stream>>>(w_ih, w_hh, pack);
  lstm_scan_kernel<<<B_ / 16, 256, SMEM_BYTES, stream>>>(x, b_ih, b_hh, w_out,
                                                         b_out, pack, out);
}
